// Gemma3nAudioAttention_27719718928496
// MI455X (gfx1250) — matrix operand order repaired and checked by reading, not compiled, not run on hardware
//
#include <hip/hip_runtime.h>
#include <math.h>

// ---------------- types ----------------
typedef __attribute__((ext_vector_type(4)))  float          f32x4;
typedef __attribute__((ext_vector_type(8)))  float          v8f;
typedef __attribute__((ext_vector_type(8)))  unsigned short u16x8;
typedef __attribute__((ext_vector_type(16))) unsigned short u16x16;
typedef __attribute__((ext_vector_type(16))) __bf16         v16bf;
typedef __attribute__((ext_vector_type(4)))  int            v4i;

// ---------------- problem constants ----------------
#define HSZ   1536
#define NH    8
#define HD    192
#define BATCH 8
#define TLEN  3072
#define BT    (BATCH*TLEN)   // 24576
#define UCNT  (TLEN/12)      // 256
#define CTXN  24
#define SPAN  13

__device__ __forceinline__ unsigned short f2bf(float f) {
  unsigned int u = __float_as_uint(f);
  u += 0x7FFFu + ((u >> 16) & 1u);   // round-to-nearest-even
  return (unsigned short)(u >> 16);
}

__device__ __forceinline__ v8f wmma_bf16(u16x16 a, u16x16 b, v8f c) {
  return __builtin_amdgcn_wmma_f32_16x16x32_bf16(
      false, __builtin_bit_cast(v16bf, a),
      false, __builtin_bit_cast(v16bf, b),
      (short)0, c, false, false);
}

__device__ __forceinline__ float fast_tanh(float x) {
#if __has_builtin(__builtin_amdgcn_tanhf)
  return __builtin_amdgcn_tanhf(x);
#elif __has_builtin(__builtin_amdgcn_tanh_f32)
  return __builtin_amdgcn_tanh_f32(x);
#else
  float e = __expf(2.0f * x);
  return (e - 1.0f) / (e + 1.0f);
#endif
}

// ---------------- async global->LDS (CDNA5) ----------------
#if __has_builtin(__builtin_amdgcn_global_load_async_to_lds_b128)
#define HAVE_ASYNC 1
typedef __attribute__((address_space(1))) v4i gas_v4i;
typedef __attribute__((address_space(3))) v4i las_v4i;
__device__ __forceinline__ void async_cp16(const unsigned short* g, unsigned short* l) {
  v4i* gp = (v4i*)(void*)const_cast<unsigned short*>(g);
  v4i* lp = (v4i*)(void*)l;
  __builtin_amdgcn_global_load_async_to_lds_b128((gas_v4i*)gp, (las_v4i*)lp, 0, 0);
}
#define WAIT_ASYNC(n) asm volatile("s_wait_asynccnt %0" :: "n"(n) : "memory")
#else
#define HAVE_ASYNC 0
__device__ __forceinline__ void async_cp16(const unsigned short* g, unsigned short* l) {
  *(u16x8*)l = *(const u16x8*)g;
}
#define WAIT_ASYNC(n) do {} while (0)
#endif

// =====================================================================
// Kernel 0a: fp32 -> bf16 bulk convert (row-major)
// =====================================================================
__global__ __launch_bounds__(256) void cvt_bf16(
    const float* __restrict__ src, unsigned short* __restrict__ dst, int n8)
{
  int i = blockIdx.x * 256 + threadIdx.x;
  if (i >= n8) return;
  const float* s = src + (size_t)i * 8;
  f32x4 f0 = *(const f32x4*)s;
  f32x4 f1 = *(const f32x4*)(s + 4);
  u16x8 h;
#pragma unroll
  for (int e = 0; e < 4; ++e) { h[e] = f2bf(f0[e]); h[4 + e] = f2bf(f1[e]); }
  *(u16x8*)(dst + (size_t)i * 8) = h;
}

// =====================================================================
// Kernel 0b: weight fp32 [k][n] -> bf16 transposed [n][k]
// =====================================================================
__global__ __launch_bounds__(256) void cvt_T_bf16(
    const float* __restrict__ w, unsigned short* __restrict__ wT)
{
  int i = blockIdx.x * 256 + threadIdx.x;   // HSZ * (HSZ/8) groups
  int k  = i / (HSZ / 8);
  int ng = (i % (HSZ / 8)) * 8;
  f32x4 f0 = *(const f32x4*)(w + (size_t)k * HSZ + ng);
  f32x4 f1 = *(const f32x4*)(w + (size_t)k * HSZ + ng + 4);
#pragma unroll
  for (int e = 0; e < 4; ++e) {
    wT[(size_t)(ng + e) * HSZ + k]     = f2bf(f0[e]);
    wT[(size_t)(ng + 4 + e) * HSZ + k] = f2bf(f1[e]);
  }
}

// =====================================================================
// Kernel 1: qkv projection (all-bf16 operands), out[m,n]=sum_k xb[m,k]*wT[n,k]
// 256 threads (8 waves), tile 64(M) x 128(N), BK=64, double-buffered async
// =====================================================================
__global__ __launch_bounds__(256) void qkv_gemm(
    const unsigned short* __restrict__ xb, const unsigned short* __restrict__ wT,
    const float* __restrict__ pds, unsigned short* __restrict__ o, int scale_q)
{
  __shared__ unsigned short lA[2][64][64];    // [m][k] bf16
  __shared__ unsigned short lB[2][128][64];   // [n][k] bf16
  const int tid = threadIdx.x;
  const int l   = tid & 31;
  const int wv  = tid >> 5;
  const int m0  = blockIdx.y * 64;
  const int n0  = blockIdx.x * 128;
  const int wm  = (wv & 3) * 16;
  const int wn  = (wv >> 2) * 64;
  const int lm  = l & 15;
  const int khA = (l >> 4) * 16;   // A frag K-half select
  const int khB = (l >> 4) * 16;   // B frag K-half select
  const int row = wm + lm;

  v8f acc[4] = {{}, {}, {}, {}};

  // stage issue: 2x16B for A-tile (8KB), 4x16B for B-tile (16KB) per thread
  auto issue = [&](int kt, int bb) {
    const int k0 = kt * 64;
#pragma unroll
    for (int j = 0; j < 2; ++j) {
      int u = tid + 256 * j;
      int r = u >> 3, cs = (u & 7) * 8;
      async_cp16(xb + (size_t)(m0 + r) * HSZ + k0 + cs, &lA[bb][r][cs]);
    }
#pragma unroll
    for (int j = 0; j < 4; ++j) {
      int u = tid + 256 * j;
      int r = u >> 3, cs = (u & 7) * 8;
      async_cp16(wT + (size_t)(n0 + r) * HSZ + k0 + cs, &lB[bb][r][cs]);
    }
  };

  issue(0, 0);
  for (int kt = 0; kt < HSZ / 64; ++kt) {
    const int bb = kt & 1;
    if (kt + 1 < HSZ / 64) {
      issue(kt + 1, bb ^ 1);
      WAIT_ASYNC(6);          // the 6 newest are next-tile; current tile done
    } else {
      WAIT_ASYNC(0);
    }
    __syncthreads();

    // hoist ALL fragment loads before the WMMA chain (one dscnt wait total)
    u16x16 avq[2];
    u16x16 bvq[8];
#pragma unroll
    for (int q = 0; q < 2; ++q) {
      u16x8 a0 = *(const u16x8*)&lA[bb][row][q * 32 + khA];
      u16x8 a1 = *(const u16x8*)&lA[bb][row][q * 32 + 8 + khA];
      avq[q] = __builtin_shufflevector(a0, a1, 0,1,2,3,4,5,6,7,8,9,10,11,12,13,14,15);
#pragma unroll
      for (int i = 0; i < 4; ++i)
        bvq[q * 4 + i] = *(const u16x16*)&lB[bb][wn + i * 16 + lm][q * 32 + khB];
    }
#pragma unroll
    for (int q = 0; q < 2; ++q)
#pragma unroll
      for (int i = 0; i < 4; ++i)
        acc[i] = wmma_bf16(avq[q], bvq[q * 4 + i], acc[i]);

    __syncthreads();
  }

  // epilogue: per-column scale factor (4 per lane), then bf16 store
  float fac[4];
#pragma unroll
  for (int i = 0; i < 4; ++i) {
    fac[i] = 1.0f;
    if (scale_q) {
      int d = (n0 + wn + i * 16 + lm) % HD;
      fac[i] = 0.1041175070f * __logf(1.0f + __expf(pds[d]));  // 192^-.5/ln2 * softplus
    }
  }
#pragma unroll
  for (int i = 0; i < 4; ++i) {
#pragma unroll
    for (int v = 0; v < 8; ++v) {
      int mr = m0 + wm + v + ((l >> 4) << 3);
      int nc = n0 + wn + i * 16 + lm;
      o[(size_t)mr * HSZ + nc] = f2bf(acc[i][v] * fac[i]);
    }
  }
}

// =====================================================================
// Kernel 2: sin_emb[f, :] = timing_signal(f) @ w_pos   (bf16, rows 13..15 = 0)
// =====================================================================
__global__ __launch_bounds__(256) void pos_emb(
    const float* __restrict__ wpos, unsigned short* __restrict__ sinw)
{
  int idx = blockIdx.x * 256 + threadIdx.x;
  if (idx >= 16 * HSZ) return;
  int f = idx / HSZ, j = idx % HSZ;
  if (f >= SPAN) { sinw[idx] = 0; return; }
  float pos = 12.0f - (float)f;
  const float loginc = 9.210340371976184f / 767.0f;   // log(1e4)/(768-1)
  float acc = 0.0f;
#pragma unroll 1
  for (int i = 0; i < HSZ; ++i) {
    float ts;
    if (i < 768) ts = __sinf(pos * __expf(-loginc * (float)i));
    else         ts = __cosf(pos * __expf(-loginc * (float)(i - 768)));
    acc += ts * wpos[(size_t)i * HSZ + j];
  }
  sinw[idx] = f2bf(acc);
}

// =====================================================================
// Kernel 3: local attention. grid = B*U blocks, 8 waves (one per head)
// =====================================================================
__global__ __launch_bounds__(256) void local_attn(
    const unsigned short* __restrict__ qw, const unsigned short* __restrict__ kw,
    const unsigned short* __restrict__ vw, const unsigned short* __restrict__ sinw,
    const unsigned char* __restrict__ msk, float* __restrict__ out)
{
  __shared__ float          scores[NH][16][48];   // [..][w][c:0-31 ac | 32-47 bd]
  __shared__ unsigned short probs [NH][16][32];   // bf16, padded
  __shared__ unsigned short vt    [NH][32][32];   // V^T chunk: [d_local][c]
  const int tid = threadIdx.x;
  const int l   = tid & 31;
  const int h   = tid >> 5;
  const int bu  = blockIdx.x;
  const int b   = bu >> 8;           // U = 256
  const int u   = bu & 255;
  const int t_base = u * 12 - 12;    // key time of context index 0
  const int lm  = l & 15;
  const int khA = (l >> 4) * 16;
  const int khB = (l >> 4) * 16;

  // ---------- logits: term_ac (2 tiles) + term_bd (1 tile) ----------
  v8f acc0 = {}, acc1 = {}, acc2 = {};
  const int t0 = t_base + lm;
  const int t1 = t_base + 16 + lm;
  bool v0 = (t0 >= 0) && (t0 < TLEN);
  if (v0) v0 = (msk[b * TLEN + t0] == 0);
  bool v1 = (lm < 8) && (t1 >= 0) && (t1 < TLEN);
  if (v1) v1 = (msk[b * TLEN + t1] == 0);

  const unsigned short* qrow  = qw + ((size_t)(b * TLEN + u * 12 + lm)) * HSZ + h * HD;
  const unsigned short* krow0 = kw + ((size_t)(b * TLEN + (v0 ? t0 : 0))) * HSZ + h * HD;
  const unsigned short* krow1 = kw + ((size_t)(b * TLEN + (v1 ? t1 : 0))) * HSZ + h * HD;
  const unsigned short* srow  = sinw + (size_t)lm * HSZ + h * HD;

#pragma unroll
  for (int kk = 0; kk < 6; ++kk) {
    const int k0 = kk * 32;
    u16x8 a0 = {}, a1 = {};
    if (lm < 12) {
      a0 = *(const u16x8*)(qrow + k0 + khA);
      a1 = *(const u16x8*)(qrow + k0 + 8 + khA);
    }
    u16x16 av = __builtin_shufflevector(a0, a1, 0,1,2,3,4,5,6,7,8,9,10,11,12,13,14,15);
    u16x16 b0 = {}; if (v0) b0 = *(const u16x16*)(krow0 + k0 + khB);
    u16x16 b1 = {}; if (v1) b1 = *(const u16x16*)(krow1 + k0 + khB);
    u16x16 b2 = *(const u16x16*)(srow + k0 + khB);
    acc0 = wmma_bf16(av, b0, acc0);
    acc1 = wmma_bf16(av, b1, acc1);
    acc2 = wmma_bf16(av, b2, acc2);
  }
#pragma unroll
  for (int v = 0; v < 8; ++v) {
    int r = v + ((l >> 4) << 3);
    scores[h][r][lm]      = acc0[v];
    scores[h][r][16 + lm] = acc1[v];
    scores[h][r][32 + lm] = acc2[v];
  }
  __syncthreads();

  // ---------- rel-shift + softcap + mask + softmax (one lane per row) ----------
  if (l < 16) {
    if (l < 12) {
      float p[CTXN];
      float mx = -3.0e38f;
#pragma unroll 1
      for (int c = 0; c < CTXN; ++c) {
        int t = t_base + c;
        bool cond = (c >= l) && (c <= l + 12) && (t >= 0) && (t < TLEN);
        if (cond) cond = (msk[b * TLEN + t] == 0);
        float lo = -3.0e38f;
        if (cond) {
          lo = scores[h][l][c] + scores[h][l][32 + (c - l)];  // bd[w, c-w]
          lo = fast_tanh(lo * 0.02f) * 50.0f;                 // softcap 50
        }
        p[c] = lo;
        mx = fmaxf(mx, lo);
      }
      float s = 0.0f;
#pragma unroll 1
      for (int c = 0; c < CTXN; ++c) { float e = __expf(p[c] - mx); p[c] = e; s += e; }
      float inv = 1.0f / s;
#pragma unroll 1
      for (int c = 0; c < CTXN; ++c) probs[h][l][c] = f2bf(p[c] * inv);
#pragma unroll
      for (int c = CTXN; c < 32; ++c) probs[h][l][c] = 0;
    } else {
#pragma unroll 1
      for (int c = 0; c < 32; ++c) probs[h][l][c] = 0;
    }
  }
  __syncthreads();

  // ---------- ctx = probs @ V  (d in chunks of 32) ----------
  u16x8 pa0 = *(const u16x8*)&probs[h][lm][khA];
  u16x8 pa1 = *(const u16x8*)&probs[h][lm][8 + khA];
  u16x16 pav = __builtin_shufflevector(pa0, pa1, 0,1,2,3,4,5,6,7,8,9,10,11,12,13,14,15);

  for (int ch = 0; ch < 6; ++ch) {
    const int d0 = ch * 32;
    // transpose V chunk into LDS: vt[d_local][c] (c 24..31 zero padded)
    for (int idx = l; idx < 128; idx += 32) {
      int c = idx >> 2, seg = idx & 3;
      int t = t_base + c;
      u16x8 vv = {};
      if (c < CTXN && t >= 0 && t < TLEN && msk[b * TLEN + t] == 0)
        vv = *(const u16x8*)(vw + ((size_t)(b * TLEN + t)) * HSZ + h * HD + d0 + seg * 8);
#pragma unroll
      for (int e = 0; e < 8; ++e) vt[h][seg * 8 + e][c] = vv[e];
    }
    __syncthreads();

    u16x16 bv0 = *(const u16x16*)&vt[h][lm][khB];
    u16x16 bv1 = *(const u16x16*)&vt[h][16 + lm][khB];
    v8f z0 = {}, z1 = {};
    v8f ca0 = wmma_bf16(pav, bv0, z0);
    v8f ca1 = wmma_bf16(pav, bv1, z1);

#pragma unroll
    for (int v = 0; v < 8; ++v) {
      int wr = v + ((l >> 4) << 3);
      if (wr < 12) {
        size_t t = (size_t)(b * TLEN + u * 12 + wr);
        float* op = out + (t * NH + h) * HD + d0;
        op[lm]      = ca0[v];
        op[16 + lm] = ca1[v];
      }
    }
    __syncthreads();
  }
}

// =====================================================================
extern "C" void kernel_launch(void* const* d_in, const int* in_sizes, int n_in,
                              void* d_out, int out_size, void* d_ws, size_t ws_size,
                              hipStream_t stream) {
  (void)in_sizes; (void)n_in; (void)out_size; (void)ws_size;
  const float*         x    = (const float*)d_in[0];
  const unsigned char* mask = (const unsigned char*)d_in[1];
  const float*         w_q  = (const float*)d_in[2];
  const float*         w_k  = (const float*)d_in[3];
  const float*         w_v  = (const float*)d_in[4];
  const float*         wpos = (const float*)d_in[5];
  const float*         pds  = (const float*)d_in[6];

  const size_t XE = (size_t)BT * HSZ;            // x / q / k / v element count
  const size_t WE = (size_t)HSZ * HSZ;           // weight element count
  unsigned short* xb   = (unsigned short*)d_ws;
  unsigned short* wqT  = xb  + XE;
  unsigned short* wkT  = wqT + WE;
  unsigned short* wvT  = wkT + WE;
  unsigned short* qw   = wvT + WE;
  unsigned short* kw   = qw  + XE;
  unsigned short* vw   = kw  + XE;
  unsigned short* sinw = vw  + XE;               // 16*1536 bf16

  // bf16 staging
  cvt_bf16<<<(int)(XE / 8 / 256), 256, 0, stream>>>(x, xb, (int)(XE / 8));
  cvt_T_bf16<<<(int)(WE / 8 / 256), 256, 0, stream>>>(w_q, wqT);
  cvt_T_bf16<<<(int)(WE / 8 / 256), 256, 0, stream>>>(w_k, wkT);
  cvt_T_bf16<<<(int)(WE / 8 / 256), 256, 0, stream>>>(w_v, wvT);

  dim3 gp(HSZ / 128, BT / 64);                   // 12 x 384
  qkv_gemm<<<gp, 256, 0, stream>>>(xb, wqT, pds, qw, 1);
  qkv_gemm<<<gp, 256, 0, stream>>>(xb, wkT, pds, kw, 0);
  qkv_gemm<<<gp, 256, 0, stream>>>(xb, wvT, pds, vw, 0);

  pos_emb<<<(16 * HSZ + 255) / 256, 256, 0, stream>>>(wpos, sinw);
  local_attn<<<BATCH * UCNT, 256, 0, stream>>>(qw, kw, vw, sinw, mask, (float*)d_out);
}
